// Parser_84920093376985
// MI455X (gfx1250) — matrix operand order repaired and checked by reading, not compiled, not run on hardware
//
#include <hip/hip_runtime.h>

// ---------------------------------------------------------------------------
// Problem constants
// ---------------------------------------------------------------------------
#define Bv       32
#define Lv       512
#define BL       16384          // Bv*Lv
#define D_EMB    500
#define LANG     100
#define Hdim     800
#define P_ARC    100
#define P_LAB    400
#define MLPd     500
#define N_UPOS   20
#define N_XPOS   300
#define N_ATTRS  500

#define KHO      608            // 600 padded to multiple of 32
#define N1PAD    512            // P_ARC + P_LAB = 500 -> 512
#define NAUXPAD  512            // 500 -> 512
#define N2PAD    832            // 20+300+500 = 820 -> 832

// output layout (floats, concatenated in return order)
#define OFF_ARCS   ((size_t)0)
#define OFF_LABEL  ((size_t)8404992)   // 32*512*513
#define OFF_UPOS   ((size_t)14958592)
#define OFF_XPOS   ((size_t)15286272)
#define OFF_ATTRS  ((size_t)20201472)

// workspace layout (bytes)
#define WS_APACK   ((size_t)0)                         // bf16 [16384][608]
#define WS_HIDB    ((size_t)19922944)                  // bf16 [16384][800]
#define WS_PROJARC ((size_t)46137344)                  // f32  [16384][100]
#define WS_S1      ((size_t)52690944)                  // f32  [16384]
#define WS_S2      ((size_t)52756480)                  // f32  [16384]
#define WS_W1P     ((size_t)52822016)                  // bf16 [512][608]
#define WS_WAUXP   ((size_t)53444608)                  // bf16 [512][800]
#define WS_WALLP   ((size_t)54263808)                  // bf16 [832][608]
#define WS_BIAS1   ((size_t)55275520)                  // f32  [512]
#define WS_BIASAUX ((size_t)55277568)                  // f32  [512]
#define WS_BIAS2   ((size_t)55279616)                  // f32  [832]
#define WS_ZERO_LO WS_W1P
#define WS_ZERO_HI ((size_t)55282944)

typedef float v8f  __attribute__((ext_vector_type(8)));
typedef __bf16 v16bf __attribute__((ext_vector_type(16)));

struct U4x2 { uint4 lo, hi; };
static_assert(sizeof(U4x2) == sizeof(v16bf), "frag size");

__device__ __forceinline__ unsigned short f2bf(float f) {
    unsigned int u = __builtin_bit_cast(unsigned int, f);
    u += 0x7FFFu + ((u >> 16) & 1u);            // round-to-nearest-even
    return (unsigned short)(u >> 16);
}

__device__ __forceinline__ v16bf ld_frag(const unsigned short* p0,
                                         const unsigned short* p1) {
    U4x2 t;
    t.lo = *(const uint4*)p0;
    t.hi = *(const uint4*)p1;
    return __builtin_bit_cast(v16bf, t);
}

// CDNA5 async memory->LDS copy (16B per lane), tracked by ASYNCcnt.
__device__ __forceinline__ void async_b128(unsigned lds_off,
                                           const unsigned short* gaddr) {
    asm volatile("global_load_async_to_lds_b128 %0, %1, off"
                 :: "v"(lds_off), "v"(gaddr)
                 : "memory");
}
__device__ __forceinline__ void wait_async() {
    asm volatile("s_wait_asynccnt 0x0" ::: "memory");
}

// ---------------------------------------------------------------------------
// Pack ho = [emb | lang_emb | 0pad] as bf16 [16384][608]
// ---------------------------------------------------------------------------
__global__ void pack_ho_kernel(const float* __restrict__ emb,
                               const int*   __restrict__ lang_ids,
                               const float* __restrict__ lang_table,
                               unsigned short* __restrict__ A) {
    int idx = blockIdx.x * blockDim.x + threadIdx.x;
    if (idx >= BL * KHO) return;
    int row = idx / KHO;
    int c   = idx - row * KHO;
    float v;
    if (c < D_EMB) {
        v = emb[(size_t)row * D_EMB + c];
    } else if (c < D_EMB + LANG) {
        int b = row >> 9;                       // row / Lv
        v = lang_table[lang_ids[b] * LANG + (c - D_EMB)];
    } else {
        v = 0.0f;
    }
    A[idx] = f2bf(v);
}

// f32 -> bf16 elementwise (hidden)
__global__ void pack_hidden_kernel(const float* __restrict__ src,
                                   unsigned short* __restrict__ dst, int n) {
    int idx = blockIdx.x * blockDim.x + threadIdx.x;
    if (idx < n) dst[idx] = f2bf(src[idx]);
}

// ---------------------------------------------------------------------------
// Pack a weight matrix [Nsrc][Ksrc] f32 -> bf16 rows [rowoff..) of [*, Kpad]
// ---------------------------------------------------------------------------
__global__ void pack_w_kernel(const float* __restrict__ W,
                              const float* __restrict__ bvec,
                              int Nsrc, int Ksrc, int rowoff,
                              unsigned short* __restrict__ Wp, int Kpad,
                              float* __restrict__ biasp) {
    int idx = blockIdx.x * blockDim.x + threadIdx.x;
    if (idx >= Nsrc * Kpad) return;
    int n = idx / Kpad;
    int k = idx - n * Kpad;
    Wp[(size_t)(rowoff + n) * Kpad + k] =
        (k < Ksrc) ? f2bf(W[(size_t)n * Ksrc + k]) : (unsigned short)0;
    if (k == 0) biasp[rowoff + n] = bvec[n];
}

// ---------------------------------------------------------------------------
// Tiled bf16 WMMA GEMM:  C[m,n] = epilogue( sum_k A[m,k] * Wp[n,k] + bias[n] )
//   A  : bf16 [16384][K]   (row stride == K)
//   Wp : bf16 [Npad][K]
// Block: 256 threads = 8 wave32s; tile 128x64, BK=32; each wave -> 32x32
// (2 A-frags x 2 B-frags = 4 WMMAs per K-step). Double-buffered LDS staging
// via GLOBAL_LOAD_ASYNC_TO_LDS_B128 (ASYNCcnt): tile t+1 streams in while
// tile t feeds the WMMAs; one barrier per K-step.
// mode 1: tanh; n<100 -> o0 (proj_arc f32, 100), n<500 -> o1 (labels, 400)
// mode 2: tanh; n<500 -> ob (aux_cat bf16, stride 608)
// mode 3: none; n<20 -> o0(20), n<320 -> o1(300), n<820 -> o2(500)
// ---------------------------------------------------------------------------
__global__ __launch_bounds__(256)
void wmma_gemm_kernel(const unsigned short* __restrict__ A, int K,
                      const unsigned short* __restrict__ Wp,
                      const float* __restrict__ bias, int mode,
                      float* __restrict__ o0, float* __restrict__ o1,
                      float* __restrict__ o2, unsigned short* __restrict__ ob) {
    constexpr int BM = 128, BN = 64, BK = 32, LDK = 40;  // LDK: padded LDS row
    constexpr unsigned ASZ = BM * LDK * 2u;              // bytes per A buffer
    constexpr unsigned BSZ = BN * LDK * 2u;              // bytes per B buffer
    __shared__ unsigned short As[2][BM * LDK];
    __shared__ unsigned short Bs[2][BN * LDK];

    const int tid   = threadIdx.x;
    const int lane  = tid & 31;
    const int wave  = tid >> 5;
    const int wm    = wave >> 1;          // 0..3 (rows, 32 each)
    const int wn    = wave & 1;           // 0..1 (cols, 32 each)
    const int khalf = lane >> 4;          // 0..1
    const int l16   = lane & 15;

    const int m0 = blockIdx.y * BM;
    const int n0 = blockIdx.x * BN;

    // per-thread staging assignments (fixed across K-steps)
    const int ar0 = tid >> 2;                  // A chunk rep0: rows 0..63
    const int ar1 = (tid + 256) >> 2;          // rep1: rows 64..127
    const int akc = (tid & 3) << 3;            // 8 bf16 per 16B chunk
    const int bnr = tid >> 2;                  // B rows 0..63
    const int bkc = (tid & 3) << 3;

    const unsigned asb = (unsigned)(size_t)(void*)&As[0][0];
    const unsigned bsb = (unsigned)(size_t)(void*)&Bs[0][0];
    const unsigned offA0 = (unsigned)(ar0 * LDK + akc) * 2u;
    const unsigned offA1 = (unsigned)(ar1 * LDK + akc) * 2u;
    const unsigned offB  = (unsigned)(bnr * LDK + bkc) * 2u;

    const unsigned short* gA0 = A + (size_t)(m0 + ar0) * K + akc;
    const unsigned short* gA1 = A + (size_t)(m0 + ar1) * K + akc;
    const unsigned short* gB  = Wp + (size_t)(n0 + bnr) * K + bkc;

    v8f acc[2][2] = {{{}, {}}, {{}, {}}};

    // preload tile 0 into buffer 0
    async_b128(asb + offA0, gA0);
    async_b128(asb + offA1, gA1);
    async_b128(bsb + offB,  gB);
    wait_async();
    __syncthreads();

    const int nT = K / BK;
    for (int t = 0; t < nT; ++t) {
        const int cur = t & 1;
        const int nxt = cur ^ 1;
        const bool more = (t + 1) < nT;

        if (more) {                            // stream tile t+1 while computing
            const int ko = (t + 1) * BK;
            async_b128(asb + nxt * ASZ + offA0, gA0 + ko);
            async_b128(asb + nxt * ASZ + offA1, gA1 + ko);
            async_b128(bsb + nxt * BSZ + offB,  gB  + ko);
        }

        const unsigned short* a0p = &As[cur][(wm * 32 + l16) * LDK + khalf * 8];
        const unsigned short* a1p = a0p + 16 * LDK;
        v16bf a0 = ld_frag(a0p, a0p + 16);
        v16bf a1 = ld_frag(a1p, a1p + 16);

        const unsigned short* b0p = &Bs[cur][(wn * 32 + l16) * LDK + khalf * 8];
        const unsigned short* b1p = b0p + 16 * LDK;
        v16bf b0 = ld_frag(b0p, b0p + 16);
        v16bf b1 = ld_frag(b1p, b1p + 16);

        acc[0][0] = __builtin_amdgcn_wmma_f32_16x16x32_bf16(
                        false, a0, false, b0, (short)0, acc[0][0], false, false);
        acc[0][1] = __builtin_amdgcn_wmma_f32_16x16x32_bf16(
                        false, a0, false, b1, (short)0, acc[0][1], false, false);
        acc[1][0] = __builtin_amdgcn_wmma_f32_16x16x32_bf16(
                        false, a1, false, b0, (short)0, acc[1][0], false, false);
        acc[1][1] = __builtin_amdgcn_wmma_f32_16x16x32_bf16(
                        false, a1, false, b1, (short)0, acc[1][1], false, false);

        if (more) {
            wait_async();                      // tile t+1 landed
            __syncthreads();                   // all readers of buf[cur] done
        }
    }

    // --- epilogue: bias + activation + routed scatter ---
    const int nbase = n0 + wn * 32;
    const float bn0 = bias[nbase + l16];
    const float bn1 = bias[nbase + 16 + l16];

    #pragma unroll
    for (int mi = 0; mi < 2; ++mi) {
        #pragma unroll
        for (int r = 0; r < 8; ++r) {
            const int m = m0 + wm * 32 + mi * 16 + khalf * 8 + r;
            #pragma unroll
            for (int nj = 0; nj < 2; ++nj) {
                const int n = nbase + nj * 16 + l16;
                float v = acc[mi][nj][r] + (nj ? bn1 : bn0);
                if (mode == 1) {
                    v = tanhf(v);
                    if (n < P_ARC)  o0[(size_t)m * P_ARC + n] = v;
                    else if (n < P_ARC + P_LAB)
                        o1[(size_t)m * P_LAB + (n - P_ARC)] = v;
                } else if (mode == 2) {
                    if (n < MLPd)
                        ob[(size_t)m * KHO + n] = f2bf(tanhf(v));
                } else {
                    if (n < N_UPOS) o0[(size_t)m * N_UPOS + n] = v;
                    else if (n < N_UPOS + N_XPOS)
                        o1[(size_t)m * N_XPOS + (n - N_UPOS)] = v;
                    else if (n < N_UPOS + N_XPOS + N_ATTRS)
                        o2[(size_t)m * N_ATTRS + (n - N_UPOS - N_XPOS)] = v;
                }
            }
        }
    }
}

// ---------------------------------------------------------------------------
// s1/s2: per-row dot of proj_arc[m,0:100] with w_head halves
// ---------------------------------------------------------------------------
__global__ void head_dots_kernel(const float* __restrict__ projarc,
                                 const float* __restrict__ w_head,
                                 float* __restrict__ s1,
                                 float* __restrict__ s2) {
    int m = blockIdx.x * blockDim.x + threadIdx.x;
    if (m >= BL) return;
    const float* row = projarc + (size_t)m * P_ARC;
    float a1 = 0.0f, a2 = 0.0f;
    for (int c = 0; c < P_ARC; ++c) {
        float t = row[c];
        a1 += t * w_head[c];
        a2 += t * w_head[P_ARC + c];
    }
    s1[m] = a1;
    s2[m] = a2;
}

// ---------------------------------------------------------------------------
// arcs[b,i,j]: j==0 -> 0 ; else s1[b,i] + s2[b,j-1] + b_head
// ---------------------------------------------------------------------------
__global__ void arcs_kernel(const float* __restrict__ s1,
                            const float* __restrict__ s2,
                            const float* __restrict__ b_head,
                            float* __restrict__ out) {
    size_t idx = (size_t)blockIdx.x * blockDim.x + threadIdx.x;
    if (idx >= (size_t)Bv * Lv * (Lv + 1)) return;
    int b   = (int)(idx / ((size_t)Lv * (Lv + 1)));
    int rem = (int)(idx - (size_t)b * Lv * (Lv + 1));
    int i   = rem / (Lv + 1);
    int j   = rem - i * (Lv + 1);
    float v = 0.0f;
    if (j != 0) v = s1[b * Lv + i] + s2[b * Lv + (j - 1)] + b_head[0];
    out[idx] = v;
}

// ---------------------------------------------------------------------------
extern "C" void kernel_launch(void* const* d_in, const int* in_sizes, int n_in,
                              void* d_out, int out_size, void* d_ws, size_t ws_size,
                              hipStream_t stream) {
    const float* emb        = (const float*)d_in[0];
    const float* hidden     = (const float*)d_in[1];
    const int*   lang_ids   = (const int*)  d_in[2];
    const float* lang_table = (const float*)d_in[3];
    const float* W_arc      = (const float*)d_in[4];
    const float* b_arc      = (const float*)d_in[5];
    const float* W_lab      = (const float*)d_in[6];
    const float* b_lab      = (const float*)d_in[7];
    const float* w_head     = (const float*)d_in[8];
    const float* b_head     = (const float*)d_in[9];
    const float* W_aux      = (const float*)d_in[10];
    const float* b_aux      = (const float*)d_in[11];
    const float* W_upos     = (const float*)d_in[12];
    const float* b_upos     = (const float*)d_in[13];
    const float* W_xpos     = (const float*)d_in[14];
    const float* b_xpos     = (const float*)d_in[15];
    const float* W_attrs    = (const float*)d_in[16];
    const float* b_attrs    = (const float*)d_in[17];

    float* out       = (float*)d_out;
    float* out_arcs  = out + OFF_ARCS;
    float* out_label = out + OFF_LABEL;
    float* out_upos  = out + OFF_UPOS;
    float* out_xpos  = out + OFF_XPOS;
    float* out_attrs = out + OFF_ATTRS;

    char* ws = (char*)d_ws;
    unsigned short* A_pack  = (unsigned short*)(ws + WS_APACK);
    unsigned short* hid_bf  = (unsigned short*)(ws + WS_HIDB);
    float*          projarc = (float*)(ws + WS_PROJARC);
    float*          s1      = (float*)(ws + WS_S1);
    float*          s2      = (float*)(ws + WS_S2);
    unsigned short* W1p     = (unsigned short*)(ws + WS_W1P);
    unsigned short* Wauxp   = (unsigned short*)(ws + WS_WAUXP);
    unsigned short* Wallp   = (unsigned short*)(ws + WS_WALLP);
    float*          bias1   = (float*)(ws + WS_BIAS1);
    float*          biasaux = (float*)(ws + WS_BIASAUX);
    float*          bias2   = (float*)(ws + WS_BIAS2);

    // zero the padded weight/bias region (pad rows & pad K must be 0)
    hipMemsetAsync(ws + WS_ZERO_LO, 0, WS_ZERO_HI - WS_ZERO_LO, stream);

    const int TPB = 256;
    auto blocks = [](long long n) { return (unsigned)((n + 255) / 256); };

    // weight packing (bf16, padded)
    pack_w_kernel<<<blocks((long long)P_ARC * KHO), TPB, 0, stream>>>(
        W_arc, b_arc, P_ARC, 600, 0, W1p, KHO, bias1);
    pack_w_kernel<<<blocks((long long)P_LAB * KHO), TPB, 0, stream>>>(
        W_lab, b_lab, P_LAB, 600, P_ARC, W1p, KHO, bias1);
    pack_w_kernel<<<blocks((long long)MLPd * Hdim), TPB, 0, stream>>>(
        W_aux, b_aux, MLPd, Hdim, 0, Wauxp, Hdim, biasaux);
    pack_w_kernel<<<blocks((long long)N_UPOS * KHO), TPB, 0, stream>>>(
        W_upos, b_upos, N_UPOS, 600, 0, Wallp, KHO, bias2);
    pack_w_kernel<<<blocks((long long)N_XPOS * KHO), TPB, 0, stream>>>(
        W_xpos, b_xpos, N_XPOS, 600, N_UPOS, Wallp, KHO, bias2);
    pack_w_kernel<<<blocks((long long)N_ATTRS * KHO), TPB, 0, stream>>>(
        W_attrs, b_attrs, N_ATTRS, 600, N_UPOS + N_XPOS, Wallp, KHO, bias2);

    // activations -> bf16
    pack_ho_kernel<<<blocks((long long)BL * KHO), TPB, 0, stream>>>(
        emb, lang_ids, lang_table, A_pack);
    pack_hidden_kernel<<<blocks((long long)BL * Hdim), TPB, 0, stream>>>(
        hidden, hid_bf, BL * Hdim);

    // GEMM 1: proj_arc (-> ws f32) + proj_label (-> out), tanh
    wmma_gemm_kernel<<<dim3(N1PAD / 64, BL / 128), TPB, 0, stream>>>(
        A_pack, KHO, W1p, bias1, 1, projarc, out_label, (float*)nullptr,
        (unsigned short*)nullptr);

    // biaffine head scores
    head_dots_kernel<<<blocks(BL), TPB, 0, stream>>>(projarc, w_head, s1, s2);
    arcs_kernel<<<blocks((long long)Bv * Lv * (Lv + 1)), TPB, 0, stream>>>(
        s1, s2, b_head, out_arcs);

    // GEMM 2: aux_hid = tanh(hidden @ W_aux^T + b) -> cols 0..499 of A_pack
    // (cols 500..607 still hold lang_emb / zeros => A_pack becomes aux_cat)
    wmma_gemm_kernel<<<dim3(NAUXPAD / 64, BL / 128), TPB, 0, stream>>>(
        hid_bf, Hdim, Wauxp, biasaux, 2, (float*)nullptr, (float*)nullptr,
        (float*)nullptr, A_pack);

    // GEMM 3: [upos | xpos | attrs] = aux_cat @ W_all^T + b (no activation)
    wmma_gemm_kernel<<<dim3(N2PAD / 64, BL / 128), TPB, 0, stream>>>(
        A_pack, KHO, Wallp, bias2, 3, out_upos, out_xpos, out_attrs,
        (unsigned short*)nullptr);

    (void)in_sizes; (void)n_in; (void)out_size; (void)ws_size;
}
